// LayerLinearMaskProbabilitiesPruneSign_83124797047558
// MI455X (gfx1250) — matrix operand order repaired and checked by reading, not compiled, not run on hardware
//
#include <hip/hip_runtime.h>
#include <hip/hip_fp16.h>

typedef __attribute__((ext_vector_type(16))) _Float16 v16h;
typedef __attribute__((ext_vector_type(8)))  float    v8f;

#define O_DIM 4096
#define I_DIM 4096
#define B_DIM 512
#define EPSF  1e-20f

// ---------------------------------------------------------------------------
// Kernel 1: Gumbel-argmax ternary mask -> masked weights, packed as split-f16
//   action = argmax_k( z_k + g_k ),  g = -log(-log(u+eps)+eps)
//   mask: 0->+1, 1->-1, 2->0 ;  wm = w*mask ;  wm ~= (float)whi + (float)wlo
// ---------------------------------------------------------------------------
__global__ __launch_bounds__(256) void mask_pack_w(
    const float* __restrict__ w,
    const float* __restrict__ basep,
    const float* __restrict__ flipp,
    const float* __restrict__ prunep,
    const float* __restrict__ noise,     // (O, I, 3)
    _Float16* __restrict__ whi,
    _Float16* __restrict__ wlo)
{
    size_t e = (size_t)blockIdx.x * blockDim.x + threadIdx.x;
    float u0 = noise[3 * e + 0];
    float u1 = noise[3 * e + 1];
    float u2 = noise[3 * e + 2];
    float g0 = -__logf(-__logf(u0 + EPSF) + EPSF);
    float g1 = -__logf(-__logf(u1 + EPSF) + EPSF);
    float g2 = -__logf(-__logf(u2 + EPSF) + EPSF);
    float z0 = basep[e]  + g0;
    float z1 = flipp[e]  + g1;
    float z2 = prunep[e] + g2;
    // first-index-wins argmax (matches jnp.argmax)
    int   a    = 0;
    float best = z0;
    if (z1 > best) { best = z1; a = 1; }
    if (z2 > best) { a = 2; }
    float m  = (a == 0) ? 1.0f : ((a == 1) ? -1.0f : 0.0f);
    float wm = w[e] * m;
    _Float16 hi = (_Float16)wm;
    _Float16 lo = (_Float16)(wm - (float)hi);
    whi[e] = hi;
    wlo[e] = lo;
}

// ---------------------------------------------------------------------------
// Kernel 2: split-f16 packing of x
// ---------------------------------------------------------------------------
__global__ __launch_bounds__(256) void pack_x(
    const float* __restrict__ x,
    _Float16* __restrict__ xhi,
    _Float16* __restrict__ xlo)
{
    size_t e = (size_t)blockIdx.x * blockDim.x + threadIdx.x;
    float v = x[e];
    _Float16 hi = (_Float16)v;
    _Float16 lo = (_Float16)(v - (float)hi);
    xhi[e] = hi;
    xlo[e] = lo;
}

// ---------------------------------------------------------------------------
// WMMA fragment loaders (wave32, 16-bit element layouts per CDNA5 ISA 7.12.2).
// 32-bit element indices (max index < 2^25) so the backend can use saddr-form
// global_load_b128 with loop-invariant voffsets + uniform scalar base updates.
// A 16x32 (MxK): lane L (m=L%16, h=L/16): v0-3 hold K=h*8..h*8+7,
//                v4-7 hold K=16+h*8..16+h*8+7  -> two contiguous 16B chunks.
// B 32x16 (KxN): lane L (n=L%16, h=L/16): v0-3 hold K=h*8..h*8+7,
//                v4-7 hold K=16+h*8..16+h*8+7  -> two contiguous 16B chunks.
// ---------------------------------------------------------------------------
__device__ __forceinline__ v16h load_a_frag(const _Float16* __restrict__ base,
                                            int idx)
{
    union { v16h v; float4 q[2]; } u;
    u.q[0] = *(const float4*)(base + idx);
    u.q[1] = *(const float4*)(base + idx + 16);
    return u.v;
}

__device__ __forceinline__ v16h load_b_frag(const _Float16* __restrict__ base,
                                            int idx)
{
    union { v16h v; float4 q[2]; } u;
    u.q[0] = *(const float4*)(base + idx);
    u.q[1] = *(const float4*)(base + idx + 16);
    return u.v;
}

__device__ __forceinline__ v8f wmma_f16(v16h a, v16h b, v8f c)
{
    return __builtin_amdgcn_wmma_f32_16x16x32_f16(
        /*neg_a=*/false, a, /*neg_b=*/false, b,
        /*c_mod=*/(short)0, c, /*reuse_a=*/false, /*reuse_b=*/false);
}

// ---------------------------------------------------------------------------
// Kernel 3: C(512x4096) = Xsplit(512x4096) * Wsplit(4096x4096)^T + bias
// Block = 256 threads (8 waves). Each wave: one 32(M) x 64(N) strip:
// 2 m-tiles x 4 n-tiles = 8 accumulator tiles. K=4096 in chunks of 32.
// Per chunk: 4 A-frag + 8 B-frag loads (24 x global_load_b128) feed 24 WMMAs
// (per tile: hi*hi + lo*hi + hi*lo). B frags reused across both m-tiles.
// ---------------------------------------------------------------------------
__global__ __launch_bounds__(256) void gemm_wmma(
    const _Float16* __restrict__ xhi,
    const _Float16* __restrict__ xlo,
    const _Float16* __restrict__ whi,
    const _Float16* __restrict__ wlo,
    const float* __restrict__ bias,
    float* __restrict__ out)
{
    const int lane = threadIdx.x & 31;
    const int wave = threadIdx.x >> 5;
    const int r    = lane & 15;
    const int h    = lane >> 4;

    const int m0 = blockIdx.y * 32;                 // batch-row tile base (2 tiles)
    const int n0 = blockIdx.x * 512 + wave * 64;    // output-col strip base (4 tiles)

    // loop-invariant per-lane fragment offsets (elements)
    int aoff[2];                 // into xhi/xlo
#pragma unroll
    for (int mt = 0; mt < 2; ++mt)
        aoff[mt] = (m0 + mt * 16 + r) * I_DIM + h * 8;

    int boff[4];                 // into whi/wlo
#pragma unroll
    for (int t = 0; t < 4; ++t)
        boff[t] = (n0 + t * 16 + r) * I_DIM + h * 8;

    v8f acc[2][4] = {};

    for (int k0 = 0; k0 < I_DIM; k0 += 32) {
        v16h a_hi0 = load_a_frag(xhi, aoff[0] + k0);
        v16h a_lo0 = load_a_frag(xlo, aoff[0] + k0);
        v16h a_hi1 = load_a_frag(xhi, aoff[1] + k0);
        v16h a_lo1 = load_a_frag(xlo, aoff[1] + k0);
#pragma unroll
        for (int t = 0; t < 4; ++t) {
            v16h b_hi = load_b_frag(whi, boff[t] + k0);
            v16h b_lo = load_b_frag(wlo, boff[t] + k0);
            acc[0][t] = wmma_f16(a_hi0, b_hi, acc[0][t]);
            acc[0][t] = wmma_f16(a_lo0, b_hi, acc[0][t]);
            acc[0][t] = wmma_f16(a_hi0, b_lo, acc[0][t]);
            acc[1][t] = wmma_f16(a_hi1, b_hi, acc[1][t]);
            acc[1][t] = wmma_f16(a_lo1, b_hi, acc[1][t]);
            acc[1][t] = wmma_f16(a_hi1, b_lo, acc[1][t]);
        }
    }

    // C/D layout: lane L -> n = L%16 ; vgpr v -> m = v + 8*(L/16)
#pragma unroll
    for (int t = 0; t < 4; ++t) {
        const int n    = n0 + t * 16 + r;
        const float bv = bias[n];
#pragma unroll
        for (int mt = 0; mt < 2; ++mt) {
#pragma unroll
            for (int v = 0; v < 8; ++v) {
                const int m = mt * 16 + v + 8 * h;
                out[(size_t)(m0 + m) * O_DIM + n] = acc[mt][t][v] + bv;
            }
        }
    }
}

// ---------------------------------------------------------------------------
extern "C" void kernel_launch(void* const* d_in, const int* in_sizes, int n_in,
                              void* d_out, int out_size, void* d_ws, size_t ws_size,
                              hipStream_t stream)
{
    const float* x      = (const float*)d_in[0];   // (512, 4096)
    const float* w      = (const float*)d_in[1];   // (4096, 4096)
    const float* bias   = (const float*)d_in[2];   // (4096,)
    const float* basep  = (const float*)d_in[3];   // (4096, 4096)
    const float* flipp  = (const float*)d_in[4];   // (4096, 4096)
    const float* prunep = (const float*)d_in[5];   // (4096, 4096)
    const float* noise  = (const float*)d_in[6];   // (4096, 4096, 3)
    float* out          = (float*)d_out;           // (512, 4096)

    // Workspace layout (all offsets 16B-aligned):
    //   [0, 32MB)        Whi  f16 (O, I)
    //   [32MB, 64MB)     Wlo  f16 (O, I)
    //   [64MB, 68MB)     Xhi  f16 (B, I)
    //   [68MB, 72MB)     Xlo  f16 (B, I)
    char* ws = (char*)d_ws;
    const size_t wbytes = (size_t)O_DIM * I_DIM * sizeof(_Float16);
    const size_t xbytes = (size_t)B_DIM * I_DIM * sizeof(_Float16);
    _Float16* whi = (_Float16*)(ws);
    _Float16* wlo = (_Float16*)(ws + wbytes);
    _Float16* xhi = (_Float16*)(ws + 2 * wbytes);
    _Float16* xlo = (_Float16*)(ws + 2 * wbytes + xbytes);

    const int nW = O_DIM * I_DIM;   // 16,777,216
    const int nX = B_DIM * I_DIM;   //  2,097,152

    mask_pack_w<<<nW / 256, 256, 0, stream>>>(w, basep, flipp, prunep, noise,
                                              whi, wlo);
    pack_x<<<nX / 256, 256, 0, stream>>>(x, xhi, xlo);

    dim3 grid(O_DIM / 512, B_DIM / 32);   // (8, 16) -> 128 blocks, 1024 waves
    gemm_wmma<<<grid, 256, 0, stream>>>(xhi, xlo, whi, wlo, bias, out);
}
